// TransformerModel_86895778332891
// MI455X (gfx1250) — matrix operand order repaired and checked by reading, not compiled, not run on hardware
//
#include <hip/hip_runtime.h>
#include <hip/hip_bf16.h>

// ---------------------------------------------------------------------------
// Model constants (from the reference)
// ---------------------------------------------------------------------------
constexpr int BB   = 2;
constexpr int LSEQ = 2048;
constexpr int DM   = 1024;
constexpr int VV   = 32000;
constexpr int NLAY = 4;
constexpr int NSTR = 2;
constexpr int DIN  = 2 * DM;      // 2048
constexpr int DSS  = 16;
constexpr int KC   = 4;
constexpr int DR   = DM / 16;     // 64
constexpr int ROWS = BB * LSEQ;   // 4096 token rows

typedef __attribute__((ext_vector_type(16))) __bf16 v16bf;
typedef __attribute__((ext_vector_type(8)))  float  v8f;

// ---------------------------------------------------------------------------
// Embedding gather: xp = emb_primary[x], xs = emb_secondary[x]
// ---------------------------------------------------------------------------
__global__ void __launch_bounds__(256)
embed_kernel(const int* __restrict__ x,
             const float* __restrict__ ep, const float* __restrict__ es,
             float* __restrict__ xp, float* __restrict__ xs)
{
    int gid = blockIdx.x * 256 + threadIdx.x;          // ROWS*DM threads
    int row = gid >> 10;                               // DM = 1024
    int col = gid & (DM - 1);
    int tok = x[row];
    xp[gid] = ep[(size_t)tok * DM + col];
    xs[gid] = es[(size_t)tok * DM + col];
}

// ---------------------------------------------------------------------------
// LayerNorm over last dim (DM=1024), one block per row
// ---------------------------------------------------------------------------
__global__ void __launch_bounds__(256)
layernorm_kernel(const float* __restrict__ X, const float* __restrict__ w,
                 const float* __restrict__ b, float* __restrict__ Y)
{
    __shared__ float ssum[256];
    __shared__ float ssq[256];
    const int row = blockIdx.x;
    const int tid = threadIdx.x;
    const float* xr = X + (size_t)row * DM;
    float s = 0.f, q = 0.f;
    for (int i = tid; i < DM; i += 256) { float v = xr[i]; s += v; q += v * v; }
    ssum[tid] = s; ssq[tid] = q;
    __syncthreads();
    for (int off = 128; off > 0; off >>= 1) {
        if (tid < off) { ssum[tid] += ssum[tid + off]; ssq[tid] += ssq[tid + off]; }
        __syncthreads();
    }
    const float mu  = ssum[0] * (1.f / DM);
    const float var = ssq[0] * (1.f / DM) - mu * mu;
    const float inv = rsqrtf(var + 1e-5f);
    for (int i = tid; i < DM; i += 256)
        Y[(size_t)row * DM + i] = (xr[i] - mu) * inv * w[i] + b[i];
}

// ---------------------------------------------------------------------------
// Generic NT GEMM on v_wmma_f32_16x16x32_bf16:
//   C[M,N] = A[M,K](lda) * B[N,K](ldb)^T  (+ bscale*bias[n]) (+ add[m,n]) (softplus)
//
// Each wave owns a 16x64 C strip: one A fragment feeds 4 WMMAs per K-step
// (4 independent v8f accumulator chains, accumulated in place).
//
// The K-loop is specialized on a *block-uniform* full/partial condition so
// the hot path (every GEMM here except xproj's single boundary block) has
// ZERO branches between WMMAs: no accumulator PHI moves, no per-iteration
// scalar guard re-evaluation. EXEC stays all-ones around every WMMA.
// A/B fragments use contiguous k = 16h + e (not the ISA table order); f32 C map per CDNA5 ISA 7.12.2.
// ---------------------------------------------------------------------------
constexpr int NT = 4;   // N sub-tiles per wave (16 cols each)

template<bool GUARD>
__device__ __forceinline__ void
gemm_core(const float* __restrict__ Arow,
          const float* __restrict__ const Brow[NT],
          const bool nok[NT], int K, v8f acc[NT])
{
    for (int k0 = 0; k0 < K; k0 += 32) {
        __builtin_prefetch(Arow + k0 + 64, 0, 1);       // global_prefetch_b8
        v16bf a;
#pragma unroll
        for (int e = 0; e < 16; ++e) a[e] = (__bf16)Arow[k0 + e];

#pragma unroll
        for (int j = 0; j < NT; ++j) {
            if (!GUARD || nok[j]) {                      // no-op in hot path
                v16bf b;
#pragma unroll
                for (int e = 0; e < 16; ++e) b[e] = (__bf16)Brow[j][k0 + e];
                acc[j] = __builtin_amdgcn_wmma_f32_16x16x32_bf16(
                    /*neg_a=*/false, a, /*neg_b=*/false, b,
                    /*c_mod=*/(short)0, acc[j], /*reuse_a=*/false, /*reuse_b=*/false);
            }
        }
    }
}

__global__ void __launch_bounds__(128)
gemm_nt_wmma(const float* __restrict__ A, int lda,
             const float* __restrict__ Bm, int ldb,
             float* __restrict__ C, int ldc,
             int M, int N, int K,
             const float* __restrict__ bias, float bscale,
             const float* __restrict__ add, int ldadd, int act)
{
    const int lane = threadIdx.x & 31;
    const int wave = threadIdx.x >> 5;
    const int h    = lane >> 4;                 // half-wave select
    const int r    = lane & 15;
    const int tile_n0 = blockIdx.x * (16 * NT);
    const int tile_m  = (blockIdx.y * 4 + wave) * 16;
    if (tile_m >= M) return;                    // wave-uniform

    // A fragment lane view: K = k0 + 16h + e
    const float* Arow = A + (size_t)(tile_m + r) * lda + 16 * h;

    // B fragment lane view per subtile: K = k0 + 16h + e, N = tile_n + r
    const float* Brow[NT];
    bool nok[NT];
#pragma unroll
    for (int j = 0; j < NT; ++j) {
        const int tn = tile_n0 + 16 * j;
        nok[j]  = (tn < N);                     // block-uniform
        Brow[j] = Bm + (size_t)((nok[j] ? tn : 0) + r) * ldb + 16 * h;
    }

    v8f acc[NT];
#pragma unroll
    for (int j = 0; j < NT; ++j) acc[j] = (v8f){0.f,0.f,0.f,0.f,0.f,0.f,0.f,0.f};

    if (tile_n0 + 16 * NT <= N) {
        gemm_core<false>(Arow, Brow, nok, K, acc);   // hot path: branch-free
    } else {
        gemm_core<true>(Arow, Brow, nok, K, acc);    // boundary block (xproj only)
    }

#pragma unroll
    for (int j = 0; j < NT; ++j) {
        if (!nok[j]) continue;                          // uniform
        const int col = tile_n0 + 16 * j + r;
        const float bv = bias ? bscale * bias[col] : 0.f;
#pragma unroll
        for (int i = 0; i < 8; ++i) {
            const int row = tile_m + 8 * h + i;         // C layout: VGPR i -> M=i (+8 hi half)
            float v = acc[j][i] + bv;
            if (add) v += add[(size_t)row * ldadd + col];
            if (act == 1) {
                // branchless stable softplus: max(v,0) + log(1 + exp(-|v|))
                v = fmaxf(v, 0.f) + __logf(1.f + __expf(-fabsf(v)));
            }
            C[(size_t)row * ldc + col] = v;
        }
    }
}

// ---------------------------------------------------------------------------
// Causal depthwise conv1d (K=4) + bias + SiLU. Reads u = XZ[:, 0:DIN] (lda=2*DIN)
// ---------------------------------------------------------------------------
__global__ void __launch_bounds__(256)
conv_silu_kernel(const float* __restrict__ XZ,
                 const float* __restrict__ cw, const float* __restrict__ cb,
                 float* __restrict__ UC)
{
    int gid = blockIdx.x * 256 + threadIdx.x;       // ROWS*DIN threads
    int d = gid & (DIN - 1);
    int m = gid >> 11;                              // DIN = 2048
    int t = m & (LSEQ - 1);
    int b = m >> 11;                                // LSEQ = 2048
    float acc = cb[d];
#pragma unroll
    for (int j = 0; j < KC; ++j) {
        int tt = t + j - (KC - 1);
        if (tt >= 0)
            acc += XZ[(size_t)(b * LSEQ + tt) * (2 * DIN) + d] * cw[d * KC + j];
    }
    float sig = 1.f / (1.f + __expf(-acc));
    UC[(size_t)m * DIN + d] = acc * sig;            // silu
}

// ---------------------------------------------------------------------------
// Selective scan. One thread per (b, d); DS=16 state in VGPRs; Bm/Cm rows
// (contiguous 32 floats at xdbl col 64) staged through LDS in 64-step chunks.
// Epilogue fused: y = (scan + u*Dp) * silu(z)
// ---------------------------------------------------------------------------
constexpr int TCH = 64;
__global__ void __launch_bounds__(256)
scan_kernel(const float* __restrict__ UC, const float* __restrict__ DT,
            const float* __restrict__ XDBL, const float* __restrict__ XZ,
            const float* __restrict__ A_log, const float* __restrict__ Dp,
            float* __restrict__ Y)
{
    __shared__ float bc[TCH * 32];                  // [i][0:16]=B, [i][16:32]=C
    const int d = blockIdx.x * 256 + threadIdx.x;   // 0..DIN-1
    const int b = blockIdx.y;

    float Av[DSS];
#pragma unroll
    for (int s = 0; s < DSS; ++s) Av[s] = -__expf(A_log[(size_t)d * DSS + s]);
    const float Dpd = Dp[d];

    float st[DSS];
#pragma unroll
    for (int s = 0; s < DSS; ++s) st[s] = 0.f;

    for (int t0 = 0; t0 < LSEQ; t0 += TCH) {
        __syncthreads();
        for (int idx = threadIdx.x; idx < TCH * 32; idx += 256) {
            int i = idx >> 5, c = idx & 31;
            bc[idx] = XDBL[(size_t)(b * LSEQ + t0 + i) * (DR + 2 * DSS) + DR + c];
        }
        __syncthreads();
        for (int i = 0; i < TCH; ++i) {
            const size_t m = (size_t)(b * LSEQ + t0 + i);
            const float u  = UC[m * DIN + d];
            const float dt = DT[m * DIN + d];
            const float z  = XZ[m * (2 * DIN) + DIN + d];
            const float du = dt * u;
            float accv = 0.f;
#pragma unroll
            for (int s = 0; s < DSS; ++s) {
                st[s] = __expf(dt * Av[s]) * st[s] + du * bc[i * 32 + s];
                accv += st[s] * bc[i * 32 + 16 + s];
            }
            const float sig = 1.f / (1.f + __expf(-z));
            Y[m * DIN + d] = (accv + u * Dpd) * (z * sig);
        }
    }
}

// ---------------------------------------------------------------------------
// SUM = XP + XS  (halves the vocab GEMM: out_p+out_s = (xp+xs)@fc_w^T + 2*fc_b)
// ---------------------------------------------------------------------------
__global__ void __launch_bounds__(256)
add_kernel(const float* __restrict__ a, const float* __restrict__ b,
           float* __restrict__ c)
{
    int gid = blockIdx.x * 256 + threadIdx.x;
    c[gid] = a[gid] + b[gid];
}

// ---------------------------------------------------------------------------
// Host orchestration
// ---------------------------------------------------------------------------
extern "C" void kernel_launch(void* const* d_in, const int* in_sizes, int n_in,
                              void* d_out, int out_size, void* d_ws, size_t ws_size,
                              hipStream_t stream)
{
    (void)in_sizes; (void)n_in; (void)out_size; (void)ws_size;

    const int*   x       = (const int*)d_in[0];
    const float* embP    = (const float*)d_in[1];
    const float* embS    = (const float*)d_in[2];
    const float* ln_w    = (const float*)d_in[3];
    const float* ln_b    = (const float*)d_in[4];
    const float* in_w    = (const float*)d_in[5];
    const float* conv_w  = (const float*)d_in[6];
    const float* conv_b  = (const float*)d_in[7];
    const float* xproj_w = (const float*)d_in[8];
    const float* dt_w    = (const float*)d_in[9];
    const float* dt_b    = (const float*)d_in[10];
    const float* A_log   = (const float*)d_in[11];
    const float* Dp      = (const float*)d_in[12];
    const float* out_w   = (const float*)d_in[13];
    const float* fc_w    = (const float*)d_in[14];
    const float* fc_b    = (const float*)d_in[15];
    float* out = (float*)d_out;

    // Workspace partition (floats); total ~253 MB
    float* ws = (float*)d_ws;
    size_t o = 0;
    auto alloc = [&](size_t n) { float* p = ws + o; o += n; return p; };
    float* XP   = alloc((size_t)ROWS * DM);
    float* XS   = alloc((size_t)ROWS * DM);
    float* XPn  = alloc((size_t)ROWS * DM);
    float* XSn  = alloc((size_t)ROWS * DM);
    float* HLN  = alloc((size_t)ROWS * DM);          // reused as SUM at the end
    float* XZ   = alloc((size_t)ROWS * 2 * DIN);
    float* UC   = alloc((size_t)ROWS * DIN);
    float* DT   = alloc((size_t)ROWS * DIN);
    float* Y    = alloc((size_t)ROWS * DIN);
    float* XDBL = alloc((size_t)ROWS * (DR + 2 * DSS));

    const dim3 blk128(128), blk256(256);
    auto gemm_grid = [](int M, int N) {
        return dim3((unsigned)((N + 16 * NT - 1) / (16 * NT)),
                    (unsigned)((M + 63) / 64));
    };

    // 1. embeddings
    embed_kernel<<<dim3((ROWS * DM) / 256), blk256, 0, stream>>>(x, embP, embS, XP, XS);

    // 2. layers
    for (int l = 0; l < NLAY; ++l) {
        for (int s = 0; s < NSTR; ++s) {
            const int ls = l * NSTR + s;
            const float* lnw = ln_w + (size_t)ls * DM;
            const float* lnb = ln_b + (size_t)ls * DM;
            const float* inw = in_w + (size_t)ls * (2 * DIN) * DM;
            const float* cw  = conv_w + (size_t)ls * DIN * KC;
            const float* cb  = conv_b + (size_t)ls * DIN;
            const float* xpw = xproj_w + (size_t)ls * (DR + 2 * DSS) * DIN;
            const float* dtw = dt_w + (size_t)ls * DIN * DR;
            const float* dtb = dt_b + (size_t)ls * DIN;
            const float* alg = A_log + (size_t)ls * DIN * DSS;
            const float* dpp = Dp + (size_t)ls * DIN;
            const float* ow  = out_w + (size_t)ls * DM * DIN;

            const float* Xin = (s == 1) ? XS : XP;
            float* Xout      = (s == 1) ? XSn : XPn;
            const float* res = (s == 1) ? XP : nullptr;   // stream 1 adds old xp

            // LN
            layernorm_kernel<<<dim3(ROWS), blk256, 0, stream>>>(Xin, lnw, lnb, HLN);
            // in-proj: XZ[4096, 4096] = HLN[4096,1024] @ in_w^T
            gemm_nt_wmma<<<gemm_grid(ROWS, 2 * DIN), blk128, 0, stream>>>(
                HLN, DM, inw, DM, XZ, 2 * DIN, ROWS, 2 * DIN, DM,
                nullptr, 0.f, nullptr, 0, 0);
            // causal conv + silu on u-half of XZ
            conv_silu_kernel<<<dim3((ROWS * DIN) / 256), blk256, 0, stream>>>(XZ, cw, cb, UC);
            // x-proj: XDBL[4096, 96] = UC @ xproj_w^T
            gemm_nt_wmma<<<gemm_grid(ROWS, DR + 2 * DSS), blk128, 0, stream>>>(
                UC, DIN, xpw, DIN, XDBL, DR + 2 * DSS, ROWS, DR + 2 * DSS, DIN,
                nullptr, 0.f, nullptr, 0, 0);
            // dt = softplus(XDBL[:, :64] @ dt_w^T + dt_b)   (fused epilogue)
            gemm_nt_wmma<<<gemm_grid(ROWS, DIN), blk128, 0, stream>>>(
                XDBL, DR + 2 * DSS, dtw, DR, DT, DIN, ROWS, DIN, DR,
                dtb, 1.f, nullptr, 0, /*softplus=*/1);
            // selective scan (+ Dp skip + silu(z) gate fused)
            scan_kernel<<<dim3(DIN / 256, BB), blk256, 0, stream>>>(
                UC, DT, XDBL, XZ, alg, dpp, Y);
            // out-proj (+ residual for stream 1)
            gemm_nt_wmma<<<gemm_grid(ROWS, DM), blk128, 0, stream>>>(
                Y, DIN, ow, DIN, Xout, DM, ROWS, DM, DIN,
                nullptr, 0.f, res, DM, 0);
        }
        // swap new/old stream buffers
        float* t;
        t = XP; XP = XPn; XPn = t;
        t = XS; XS = XSn; XSn = t;
    }

    // 3. logits: out = (xp + xs) @ fc_w^T + 2*fc_b   (single GEMM, fc_w lives in L2)
    float* SUM = HLN;
    add_kernel<<<dim3((ROWS * DM) / 256), blk256, 0, stream>>>(XP, XS, SUM);
    gemm_nt_wmma<<<gemm_grid(ROWS, VV), blk128, 0, stream>>>(
        SUM, DM, fc_w, DM, out, VV, ROWS, VV, DM,
        fc_b, 2.f, nullptr, 0, 0);
}
